// MDL_RNN_mnist_4174708212033
// MI455X (gfx1250) — matrix operand order repaired and checked by reading, not compiled, not run on hardware
//
#include <hip/hip_runtime.h>

// ---------------------------------------------------------------------------
// SNN forward on MI455X (gfx1250).
// Dominant cost: per-step [256x4096] += [256x4096]x[4096x4096] recurrent GEMM
// in bf16 WMMA (f32 accum). Wr (32MB bf16) + all state stay resident in the
// 192MB L2 -> compute bound on v_wmma_f32_16x16x32_bf16.
// Global->LDS staging uses the CDNA5 Tensor Data Mover (tensor_load_to_lds,
// TENSORcnt) with double-buffered LDS tiles; 32x64 wave tiles cut LDS reads
// to 1.5 ds_load_b128 per WMMA.
// ---------------------------------------------------------------------------

typedef __bf16 bf16;
typedef bf16  v16bf __attribute__((ext_vector_type(16)));
typedef float v8f   __attribute__((ext_vector_type(8)));
typedef unsigned int u32x4 __attribute__((ext_vector_type(4)));
typedef int          i32x8 __attribute__((ext_vector_type(8)));
typedef int          i32x4 __attribute__((ext_vector_type(4)));

#define Hn   4096
#define Dn   784
#define DPn  800      // D padded to multiple of 32
#define Bn   256
#define Pn   128
#define On   10
#define Tn   50

#define C_LS   0.8187307530779818f   // exp(-DT/TAU_S), DT=0.002, TAU_S=0.01
#define C_LM   0.9048374180359595f   // exp(-DT/TAU_M)
#define C_LD   0.9355069850316178f   // exp(-DT/TAU_D)
#define C_DT   0.002f
#define C_DTD  0.06666666666666667f  // DT/TAU_D
#define C_TREF 0.01f                 // 5*DT
#define C_VTHR 1.0f

__device__ __forceinline__ unsigned hash3(unsigned t, unsigned b, unsigned d) {
  unsigned x = t * 0x9E3779B1u + b * 0x85EBCA77u + d * 0xC2B2AE3Du + 0x27220A95u;
  x ^= x >> 16; x *= 0x7FEB352Du;
  x ^= x >> 15; x *= 0x846CA68Bu;
  x ^= x >> 16;
  return x;
}

// A fragment (16x32 bf16, MxK): lane<16 -> K 0..15; lane>=16 -> K 16..31.
__device__ __forceinline__ v16bf frag_a(const bf16* tile, int row, int lane) {
  const int m  = lane & 15;
  const int kb = (lane < 16) ? 0 : 16;
  const bf16* p = tile + (row + m) * 32 + kb;
  v16bf f;
  ((uint4*)&f)[0] = *(const uint4*)(p);
  ((uint4*)&f)[1] = *(const uint4*)(p + 8);
  return f;
}

// B fragment (32x16 bf16, KxN) from [N][K] row-major tile: lane reads 16
// contiguous K values of its column -> one aligned 32-byte LDS read.
__device__ __forceinline__ v16bf frag_b(const bf16* tile, int col, int lane) {
  const int n  = lane & 15;
  const int kb = (lane < 16) ? 0 : 16;
  return *(const v16bf*)(tile + (col + n) * 32 + kb);
}

__device__ __forceinline__ v8f wmma_bf16(v16bf a, v16bf b, v8f c) {
  return __builtin_amdgcn_wmma_f32_16x16x32_bf16(false, a, false, b,
                                                 (short)0, c, false, false);
}

__device__ __forceinline__ unsigned lds_off(const void* p) {
  // generic pointer to LDS: low 32 bits are the group-segment byte offset
  return (unsigned)(unsigned long long)(uintptr_t)p;
}

// ---------------------------------------------------------------------------
// TDM: DMA a 2D tile (rows x 32 bf16, row-major packed) from a row-major
// tensor with stride `stride_elems` into LDS at byte offset `lo`.
// D# layout per CDNA5 ISA ch.8 (CDNA5 ISA): group0 = flags/lds/gaddr,
// group1 = data_size|dims|strides, groups 2&3 unused (2D tensor).
// This toolchain exposes the 6-arg builtin: (g0, g1, g2, g3, g4, cpol).
// ---------------------------------------------------------------------------
__device__ __forceinline__ void tdm_load_2d(unsigned lo, const bf16* gptr,
                                            unsigned rows, unsigned stride_elems) {
  unsigned long long ga = (unsigned long long)(uintptr_t)gptr;
  u32x4 g0;
  g0[0] = 1u;                                            // count=1, user flags 0
  g0[1] = lo;                                            // lds_addr
  g0[2] = (unsigned)(ga & 0xFFFFFFFFu);                  // global_addr[31:0]
  g0[3] = (unsigned)((ga >> 32) & 0x1FFFFFFu) | (2u << 30); // ga[56:32] | type=2
  i32x8 g1;
  g1[0] = (int)(1u << 16);          // workgroup_mask=0, data_size=1 (2 bytes)
  g1[1] = (int)(32u << 16);         // tensor_dim0 = 32  (bits 63:48)
  g1[2] = (int)(rows << 16);        // tensor_dim0 hi=0, tensor_dim1 lo = rows
  g1[3] = (int)(32u << 16);         // tensor_dim1 hi=0, tile_dim0 = 32
  g1[4] = (int)rows;                // tile_dim1 = rows, tile_dim2 = 0
  g1[5] = (int)stride_elems;        // tensor_dim0_stride[31:0]
  g1[6] = 0;                        // stride hi, tensor_dim1_stride lo
  g1[7] = 0;
  i32x4 z4 = {0, 0, 0, 0};
  i32x8 z8 = {0, 0, 0, 0, 0, 0, 0, 0};
  __builtin_amdgcn_tensor_load_to_lds(g0, g1, z4, z4, z8, 0);
}

// ---------------------------------------------------------------------------
// Spike tile: As[64][32] for batch rows bm.., input cols kc..kc+31.
// Branchless: float4 loads from a clamped base (always in-bounds), per-element
// select for the zero-padded tail -> no exec-mask branches, no per-load waits.
// ---------------------------------------------------------------------------
__device__ __forceinline__ void gen_spikes(bf16* As, const float* __restrict__ x,
                                           int t, int bm, int kc, int tid) {
  const int rr = tid >> 1, cc0 = (tid & 1) * 16;
  const int bglob = bm + rr;
  const int base = kc + cc0;
  const int bcl  = (base > (Dn - 16)) ? (Dn - 16) : base;  // keep loads in-bounds
  const float4* xp = (const float4*)(x + (size_t)bglob * Dn + bcl);
  alignas(16) float xv[16];
  ((float4*)xv)[0] = xp[0];
  ((float4*)xv)[1] = xp[1];
  ((float4*)xv)[2] = xp[2];
  ((float4*)xv)[3] = xp[3];
  alignas(32) bf16 tmp[16];
#pragma unroll
  for (int c = 0; c < 16; ++c) {
    const int d = base + c;
    // when base was clamped, all d >= Dn here -> p forced to 0 (loads ignored)
    float p = (d < Dn) ? fminf(fmaxf(xv[c], 0.f), 1.f) : 0.f;
    float u = (float)hash3((unsigned)t, (unsigned)bglob, (unsigned)d) *
              2.3283064365386963e-10f;
    tmp[c] = (u < p) ? (bf16)1.0f : (bf16)0.0f;
  }
  *(uint4*)&As[rr * 32 + cc0]     = ((const uint4*)tmp)[0];
  *(uint4*)&As[rr * 32 + cc0 + 8] = ((const uint4*)tmp)[1];
}

// ---------------------------------------------------------------------------
// One timestep. Block tile 64M x 128N, 128 threads = 4 waves (2x2), each wave
// a 32x64 subtile -> 8 acc, 8 WMMA per K-chunk vs 12 ds_load_b128.
// Global->LDS via TDM (wave 0 issues; s_wait_tensorcnt + barrier publishes).
// r ping-pongs between bf16 buffers -> no cross-WG races.
// Grid: (256/64) x (4096/128) = 128 WGs.
// ---------------------------------------------------------------------------
__global__ __launch_bounds__(128) void step_k(const float* __restrict__ x,
                                              const bf16* __restrict__ WinB,
                                              const bf16* __restrict__ WrB,
                                              const bf16* __restrict__ rbf_in,
                                              bf16* __restrict__ rbf_out,
                                              float* __restrict__ I,
                                              float* __restrict__ mem,
                                              float* __restrict__ s,
                                              float* __restrict__ r,
                                              float* __restrict__ tlast,
                                              float* __restrict__ rm,
                                              int t) {
  __shared__ __align__(64) bf16 As[2][64 * 32];
  __shared__ __align__(64) bf16 Bs[2][128 * 32];
  const int tid = threadIdx.x, lane = tid & 31, wid = tid >> 5;
  const int wm = (wid & 1) * 32;            // wave M offset (2 waves in M)
  const int wn = (wid >> 1) * 64;           // wave N offset (2 waves in N)
  const int bm = (blockIdx.x & 3) * 64;     // batch tile
  const int bn = (blockIdx.x >> 2) * 128;   // hidden tile
  v8f acc[2][4] = {};

  const unsigned aoff[2] = { lds_off(&As[0][0]), lds_off(&As[1][0]) };
  const unsigned boff[2] = { lds_off(&Bs[0][0]), lds_off(&Bs[1][0]) };

#define MMA_TILE(buf)                                                  \
  do {                                                                 \
    v16bf a0 = frag_a(As[buf], wm, lane);                              \
    v16bf a1 = frag_a(As[buf], wm + 16, lane);                         \
    _Pragma("unroll")                                                  \
    for (int j = 0; j < 4; ++j) {                                      \
      v16bf b = frag_b(Bs[buf], wn + j * 16, lane);                    \
      acc[0][j] = wmma_bf16(a0, b, acc[0][j]);                         \
      acc[1][j] = wmma_bf16(a1, b, acc[1][j]);                         \
    }                                                                  \
  } while (0)

  // ---- Part 1: spikes @ Win^T, K = 800 (25 chunks) ----
  gen_spikes(As[0], x, t, bm, 0, tid);
  if (wid == 0) tdm_load_2d(boff[0], WinB + (size_t)bn * DPn, 128, DPn);
  for (int i = 0; i < 24; ++i) {
    const int cb = i & 1, nb = cb ^ 1;
    gen_spikes(As[nb], x, t, bm, (i + 1) * 32, tid);
    if (wid == 0) {
      tdm_load_2d(boff[nb], WinB + (size_t)bn * DPn + (i + 1) * 32, 128, DPn);
      __builtin_amdgcn_s_wait_tensorcnt(1);   // chunk i's TDM done
    }
    __syncthreads();                          // publish chunk i
    MMA_TILE(cb);
    __syncthreads();                          // release buffers
  }
  if (wid == 0) __builtin_amdgcn_s_wait_tensorcnt(0);
  __syncthreads();
  MMA_TILE(0);                                // chunk 24 (24&1 == 0)
  __syncthreads();

  // ---- Part 2: r @ Wr^T, K = 4096 (128 chunks) ----
  if (wid == 0) {
    tdm_load_2d(aoff[0], rbf_in + (size_t)bm * Hn, 64, Hn);
    tdm_load_2d(boff[0], WrB + (size_t)bn * Hn, 128, Hn);
  }
  for (int i = 0; i < 127; ++i) {
    const int cb = i & 1, nb = cb ^ 1;
    if (wid == 0) {
      tdm_load_2d(aoff[nb], rbf_in + (size_t)bm * Hn + (i + 1) * 32, 64, Hn);
      tdm_load_2d(boff[nb], WrB + (size_t)bn * Hn + (i + 1) * 32, 128, Hn);
      __builtin_amdgcn_s_wait_tensorcnt(2);   // chunk i's two TDM ops done
    }
    __syncthreads();
    MMA_TILE(cb);
    __syncthreads();
  }
  if (wid == 0) __builtin_amdgcn_s_wait_tensorcnt(0);
  __syncthreads();
  MMA_TILE(1);                                // chunk 127 (127&1 == 1)

  // ---- Fused LIF neuron update (each lane owns its C elements) ----
  const float tnow = C_DT * (float)t;
  const int mo = (lane >= 16) ? 8 : 0, nn = lane & 15;
#pragma unroll
  for (int i = 0; i < 2; ++i)
#pragma unroll
    for (int j = 0; j < 4; ++j)
#pragma unroll
      for (int e = 0; e < 8; ++e) {
        const int row = bm + wm + i * 16 + e + mo;
        const int col = bn + wn + j * 16 + nn;
        const size_t idx = (size_t)row * Hn + col;
        float Iv  = C_LS * I[idx] + acc[i][j][e];
        float tl  = tlast[idx];
        float sv  = s[idx];
        float refr = (tnow > tl + C_TREF) ? 1.f : 0.f;
        float mv  = refr * (C_LM * mem[idx] + (1.f - C_LM) * Iv) * (1.f - sv);
        float rv  = C_LD * r[idx] + C_DTD * sv;
        float sn  = (mv - C_VTHR > 0.f) ? 1.f : 0.f;
        float tln = tl + (tnow - tl) * sn;
        I[idx] = Iv; mem[idx] = mv; s[idx] = sn; r[idx] = rv; tlast[idx] = tln;
        rm[idx] = fmaxf(rm[idx], rv);
        rbf_out[idx] = (bf16)rv;
      }
#undef MMA_TILE
}

// ---------------------------------------------------------------------------
// Prep: Win f32[H][784] -> bf16[H][800] (zero padded)
// ---------------------------------------------------------------------------
__global__ __launch_bounds__(256) void prep_win_k(const float* __restrict__ Win,
                                                  bf16* __restrict__ WinB) {
  int idx = blockIdx.x * 256 + threadIdx.x;
  if (idx >= Hn * DPn) return;
  int h = idx / DPn, d = idx - h * DPn;
  WinB[idx] = (d < Dn) ? (bf16)Win[h * Dn + d] : (bf16)0.0f;
}

// Prep: lp[h][p] = l[p]*pin[h][p] (bf16), poutB = bf16(pout)
__global__ __launch_bounds__(256) void prep_lp_k(const float* __restrict__ pin,
                                                 const float* __restrict__ pout,
                                                 const float* __restrict__ l,
                                                 bf16* __restrict__ lpB,
                                                 bf16* __restrict__ poutB) {
  int idx = blockIdx.x * 256 + threadIdx.x;
  if (idx >= Hn * Pn) return;
  int p = idx & (Pn - 1);
  lpB[idx]   = (bf16)(l[p] * pin[idx]);
  poutB[idx] = (bf16)pout[idx];
}

__global__ __launch_bounds__(256) void init_state_k(float* I, float* mem, float* s,
                                                    float* r, float* tlast, float* rm,
                                                    bf16* rbf0) {
  int idx = blockIdx.x * 256 + threadIdx.x;
  if (idx >= Bn * Hn) return;
  I[idx] = 0.f; mem[idx] = 0.f; s[idx] = 0.f; r[idx] = 0.f;
  tlast[idx] = -1.f; rm[idx] = 0.f; rbf0[idx] = (bf16)0.0f;
}

// ---------------------------------------------------------------------------
// Wr = lp @ pout^T  (4096x4096, K=128), output bf16. 64x64 tile, 4 waves.
// One-time cost (~4.3 GFLOP) — plain double-barrier pipeline is fine.
// ---------------------------------------------------------------------------
__global__ __launch_bounds__(128) void wr_gemm_k(const bf16* __restrict__ lpB,
                                                 const bf16* __restrict__ poutB,
                                                 bf16* __restrict__ WrB) {
  __shared__ __align__(64) bf16 As[64 * 32];
  __shared__ __align__(64) bf16 Bs[64 * 32];
  const int tid = threadIdx.x, lane = tid & 31, wid = tid >> 5;
  const int wm = (wid & 1) * 32, wn = (wid >> 1) * 32;
  const int bm = (blockIdx.x & 63) * 64, bn = (blockIdx.x >> 6) * 64;
  const int rr = tid >> 1, cc0 = (tid & 1) * 16;
  v8f acc[2][2] = {};

  for (int kc = 0; kc < Pn; kc += 32) {
    *(uint4*)&As[rr * 32 + cc0] = *(const uint4*)&lpB[(size_t)(bm + rr) * Pn + kc + cc0];
    *(uint4*)&Bs[rr * 32 + cc0] = *(const uint4*)&poutB[(size_t)(bn + rr) * Pn + kc + cc0];
    __syncthreads();
    v16bf a0 = frag_a(As, wm, lane),      a1 = frag_a(As, wm + 16, lane);
    v16bf b0 = frag_b(Bs, wn, lane),      b1 = frag_b(Bs, wn + 16, lane);
    acc[0][0] = wmma_bf16(a0, b0, acc[0][0]);
    acc[0][1] = wmma_bf16(a0, b1, acc[0][1]);
    acc[1][0] = wmma_bf16(a1, b0, acc[1][0]);
    acc[1][1] = wmma_bf16(a1, b1, acc[1][1]);
    __syncthreads();
  }
  const int mo = (lane >= 16) ? 8 : 0, n = lane & 15;
#pragma unroll
  for (int i = 0; i < 2; ++i)
#pragma unroll
    for (int j = 0; j < 2; ++j)
#pragma unroll
      for (int e = 0; e < 8; ++e) {
        int row = bm + wm + i * 16 + e + mo;
        int col = bn + wn + j * 16 + n;
        WrB[(size_t)row * Hn + col] = (bf16)acc[i][j][e];
      }
}

// ---------------------------------------------------------------------------
// out[b][o] = softmax(rm[b] @ Wout^T)  — tiny, one WG per batch row
// ---------------------------------------------------------------------------
__global__ __launch_bounds__(128) void out_k(const float* __restrict__ rm,
                                             const float* __restrict__ Wout,
                                             float* __restrict__ out) {
  __shared__ float sred[On][128];
  const int b = blockIdx.x, tid = threadIdx.x;
  float acc[On];
#pragma unroll
  for (int o = 0; o < On; ++o) acc[o] = 0.f;
  for (int h = tid; h < Hn; h += 128) {
    float v = rm[(size_t)b * Hn + h];
#pragma unroll
    for (int o = 0; o < On; ++o) acc[o] += v * Wout[o * Hn + h];
  }
#pragma unroll
  for (int o = 0; o < On; ++o) sred[o][tid] = acc[o];
  __syncthreads();
  for (int off = 64; off > 0; off >>= 1) {
    if (tid < off)
#pragma unroll
      for (int o = 0; o < On; ++o) sred[o][tid] += sred[o][tid + off];
    __syncthreads();
  }
  if (tid == 0) {
    float m = -1e30f;
    for (int o = 0; o < On; ++o) m = fmaxf(m, sred[o][0]);
    float e[On], sum = 0.f;
    for (int o = 0; o < On; ++o) { e[o] = __expf(sred[o][0] - m); sum += e[o]; }
    for (int o = 0; o < On; ++o) out[b * On + o] = e[o] / sum;
  }
}

// ---------------------------------------------------------------------------
extern "C" void kernel_launch(void* const* d_in, const int* in_sizes, int n_in,
                              void* d_out, int out_size, void* d_ws, size_t ws_size,
                              hipStream_t stream) {
  const float* x    = (const float*)d_in[0];   // [256][784]
  const float* Win  = (const float*)d_in[1];   // [4096][784]
  const float* Wout = (const float*)d_in[2];   // [10][4096]
  const float* pin  = (const float*)d_in[3];   // [4096][128]
  const float* pout = (const float*)d_in[4];   // [4096][128]
  const float* l    = (const float*)d_in[5];   // [128]
  float* out = (float*)d_out;

  char* ws = (char*)d_ws;
  size_t off = 0;
  bf16* WinB  = (bf16*)(ws + off); off += (size_t)Hn * DPn * 2;
  bf16* lpB   = (bf16*)(ws + off); off += (size_t)Hn * Pn * 2;
  bf16* poutB = (bf16*)(ws + off); off += (size_t)Hn * Pn * 2;
  bf16* WrB   = (bf16*)(ws + off); off += (size_t)Hn * Hn * 2;
  bf16* rbf0  = (bf16*)(ws + off); off += (size_t)Bn * Hn * 2;
  bf16* rbf1  = (bf16*)(ws + off); off += (size_t)Bn * Hn * 2;
  float* I    = (float*)(ws + off); off += (size_t)Bn * Hn * 4;
  float* mem  = (float*)(ws + off); off += (size_t)Bn * Hn * 4;
  float* s    = (float*)(ws + off); off += (size_t)Bn * Hn * 4;
  float* r    = (float*)(ws + off); off += (size_t)Bn * Hn * 4;
  float* tl   = (float*)(ws + off); off += (size_t)Bn * Hn * 4;
  float* rm   = (float*)(ws + off); off += (size_t)Bn * Hn * 4;

  prep_win_k<<<(Hn * DPn + 255) / 256, 256, 0, stream>>>(Win, WinB);
  prep_lp_k<<<(Hn * Pn + 255) / 256, 256, 0, stream>>>(pin, pout, l, lpB, poutB);
  wr_gemm_k<<<64 * 64, 128, 0, stream>>>(lpB, poutB, WrB);
  init_state_k<<<(Bn * Hn + 255) / 256, 256, 0, stream>>>(I, mem, s, r, tl, rm, rbf0);

  for (int t = 0; t < Tn; ++t) {
    bf16* rin  = (t & 1) ? rbf1 : rbf0;
    bf16* rout = (t & 1) ? rbf0 : rbf1;
    step_k<<<(Bn / 64) * (Hn / 128), 128, 0, stream>>>(x, WinB, WrB, rin, rout,
                                                       I, mem, s, r, tl, rm, t);
  }

  out_k<<<Bn, 128, 0, stream>>>(rm, Wout, out);
}
